// QuantizedSNN_12532714570358
// MI455X (gfx1250) — matrix operand order repaired and checked by reading, not compiled, not run on hardware
//
#include <hip/hip_runtime.h>

typedef __attribute__((ext_vector_type(16))) _Float16 v16h;
typedef __attribute__((ext_vector_type(8)))  _Float16 v8h;
typedef __attribute__((ext_vector_type(8)))  float    v8f;
typedef __attribute__((ext_vector_type(4)))  int      v4i;

#define B_TOT   8192
#define NI      784
#define NIP     800           // K padded to multiple of 32
#define NH      1000
#define NO      10
#define T_STEPS 25
#define BT      32            // batch rows per workgroup
#define BETA    0.95f
#define THR     1.0f
#define Q_LO    (-1.0f)
#define Q_STEP  (2.2f / 255.0f)
#define Q_MAXI  255.0f

// LDS layout (bytes). Total 291,840 <= 320 KB per WGP.
#define OFF_CUR1 0        // [32][1024] f16  (cur1 tile, cols >=1000 unused)
#define OFF_MEM1 65536    // [32][1000] f32  (membrane state, f32 precision)
#define OFF_XH   65536    // overlay during phase 1: [32][800] f16 x-tile
#define OFF_SPK1 193536   // [32][1024] f16  (spikes as 0/1 f16, K padded with zeros)
#define OFF_W2H  259072   // [16][1024] f16  (W2 padded: rows 10..15 = 0, cols 1000..1023 = 0)
#define LDS_BYTES 291840

// workspace layout (bytes): w1h f16 [1000][800] @ 0 ; xh f16 [8192][800] @ 1,600,000
#define WS_W1H_OFF 0
#define WS_XH_OFF  1600000

#if defined(__has_builtin)
#if __has_builtin(__builtin_amdgcn_global_load_async_to_lds_b128) && \
    __has_builtin(__builtin_amdgcn_s_wait_asynccnt)
#define HAVE_ASYNC_LDS 1
#endif
#endif
#ifndef HAVE_ASYNC_LDS
#define HAVE_ASYNC_LDS 0
#endif

#if HAVE_ASYNC_LDS
// Builtin signature (from hipcc diagnostic): param0 = AS1 int4* (global src),
// param1 = AS3 int4* (LDS dst), then two imm ints (offset, cpol).
#define GPTR(p) ((__attribute__((address_space(1))) v4i*)(uintptr_t)(p))
#define LPTR(p) ((__attribute__((address_space(3))) v4i*)(uintptr_t)(uint32_t)(uintptr_t)(p))
#endif

// ---------- prep: f32 -> f16 with K padding 784 -> 800 (memory-bound, ~43 MB) ----------
__global__ void __launch_bounds__(256)
prep_f16(const float* __restrict__ x, const float* __restrict__ W1,
         _Float16* __restrict__ w1h, _Float16* __restrict__ xh)
{
    const size_t stride = (size_t)gridDim.x * blockDim.x;
    size_t i0 = (size_t)blockIdx.x * blockDim.x + threadIdx.x;
    for (size_t idx = i0; idx < (size_t)NH * NIP; idx += stride) {
        int r = (int)(idx / NIP), k = (int)(idx - (size_t)r * NIP);
        w1h[idx] = (_Float16)((k < NI) ? W1[(size_t)r * NI + k] : 0.0f);
    }
    for (size_t idx = i0; idx < (size_t)B_TOT * NIP; idx += stride) {
        int r = (int)(idx / NIP), k = (int)(idx - (size_t)r * NIP);
        xh[idx] = (_Float16)((k < NI) ? x[(size_t)r * NI + k] : 0.0f);
    }
}

// ---------- fused SNN: one block owns a 32-row batch tile, all state in LDS ----------
__global__ void __launch_bounds__(256, 1)
snn_fused_wmma(const _Float16* __restrict__ xh_ws, const _Float16* __restrict__ w1h,
               const float* __restrict__ b1, const float* __restrict__ W2,
               const float* __restrict__ b2, float* __restrict__ out)
{
    extern __shared__ __align__(16) char smem[];
    _Float16* cur1h = (_Float16*)(smem + OFF_CUR1);
    float*    mem1  = (float*)   (smem + OFF_MEM1);
    _Float16* xh    = (_Float16*)(smem + OFF_XH);
    _Float16* spk1h = (_Float16*)(smem + OFF_SPK1);
    _Float16* w2h   = (_Float16*)(smem + OFF_W2H);

    const int tid  = threadIdx.x;
    const int wave = tid >> 5;          // 8 waves of 32
    const int lane = tid & 31;
    const int l16  = lane & 15;
    const int hh   = lane >> 4;         // K-half selector per WMMA operand layout
    const int b0   = blockIdx.x * BT;   // global batch row base

    // ---------- Phase 0: stage x tile (f16, pre-padded) + W2 (f16 B-operand) ----------
    {
        const v8h* src = (const v8h*)(xh_ws + (size_t)b0 * NIP);   // 51,200 B contiguous
#if HAVE_ASYNC_LDS
        v8h* dst = (v8h*)xh;
        for (int c = tid; c < BT * NIP / 8; c += 256) {
            __builtin_amdgcn_global_load_async_to_lds_b128(
                GPTR(src + c), LPTR(dst + c), 0, 0);
        }
        __builtin_amdgcn_s_wait_asynccnt(0);
#else
        for (int c = tid; c < BT * NIP / 8; c += 256)
            ((v8h*)xh)[c] = src[c];
#endif
    }
    for (int idx = tid; idx < 16 * 1024; idx += 256) {
        int o = idx >> 10, k = idx & 1023;
        float v = (o < NO && k < NH) ? W2[o * NH + k] : 0.0f;
        w2h[idx] = (_Float16)v;
    }
    __syncthreads();

    // ---------- Phase 1: cur1 = x @ W1^T + b1 via v_wmma_f32_16x16x32_f16 ----------
    // 2 M-tiles x 63 N-tiles = 126 output tiles round-robined over 8 waves.
    for (int t = wave; t < 126; t += 8) {
        const int mt = t & 1, nt = t >> 1;
        const int mbase = mt * 16, nbase = nt * 16;
        const int n  = nbase + l16;            // hidden unit (B column / D column)
        const int nc = (n < NH) ? n : NH - 1;  // clamp for safe loads; guarded at store
        const int arow = mbase + l16;          // A row for this lane
        const _Float16* wrow = &w1h[(size_t)nc * NIP + 16 * hh];  // this lane's K stream
        v8f acc = {};
        for (int kk = 0; kk < NI; kk += 32) {
            // A operand: row arow, K = kk+16h .. +15 (ds_load_b128 x2)
            v8h alo = *(const v8h*)&xh[arow * NIP + kk + 16 * hh];
            v8h ahi = *(const v8h*)&xh[arow * NIP + kk + 16 * hh + 8];
            // B operand: column n, K = kk+16h .. +15, pre-converted & padded -> branchless
            v8h blo = *(const v8h*)(wrow + kk);
            v8h bhi = *(const v8h*)(wrow + kk + 8);
            __builtin_prefetch(wrow + kk + 128, 0, 1);   // global_prefetch_b8 ahead on W1h
            v16h a, bb;
            #pragma unroll
            for (int i = 0; i < 8; ++i) {
                a[i] = alo[i];  a[i + 8] = ahi[i];
                bb[i] = blo[i]; bb[i + 8] = bhi[i];
            }
            acc = __builtin_amdgcn_wmma_f32_16x16x32_f16(false, a, false, bb,
                                                         (short)0, acc, false, false);
        }
        if (n < NH) {   // epilogue: +b1, store f16 (D: lane -> col n, rows mbase+8h+r)
            const float bv = b1[n];
            #pragma unroll
            for (int r = 0; r < 8; ++r)
                cur1h[(mbase + 8 * hh + r) * 1024 + n] = (_Float16)(acc[r] + bv);
        }
    }
    __syncthreads();

    // ---------- init state (mem1 region overlaid xh during phase 1) ----------
    for (int idx = tid; idx < BT * NH; idx += 256) mem1[idx] = 0.0f;
    for (int idx = tid; idx < BT * 1024; idx += 256) spk1h[idx] = (_Float16)0.0f;
    float mem2r[8];
    #pragma unroll
    for (int r = 0; r < 8; ++r) mem2r[r] = 0.0f;
    const int   oo  = l16;                              // output neuron (waves 0/1)
    const float b2v = (oo < NO) ? b2[oo] : 0.0f;
    __syncthreads();

    // ---------- Phase 2: 25 timesteps, fully LDS-resident ----------
    for (int t = 0; t < T_STEPS; ++t) {
        // (a) LIF layer 1: mem1 update + spike write; incremental indexing (no int div)
        {
            int n = tid, idx = tid, c = tid;    // c indexes the 1024-stride f16 buffers
            #pragma unroll 1
            for (int it = 0; it < BT * NH / 256; ++it) {
                float m     = mem1[idx];
                float cur   = (float)cur1h[c];
                float reset = (m > THR) ? 1.0f : 0.0f;
                float v     = BETA * m + cur - reset * THR;
                float q     = (v - Q_LO) * (1.0f / Q_STEP);
                q = fminf(fmaxf(q, 0.0f), Q_MAXI);
                q = Q_LO + rintf(q) * Q_STEP;           // RNE round, matches jnp.round
                mem1[idx]   = q;
                spk1h[c]    = (q > THR) ? (_Float16)1.0f : (_Float16)0.0f;
                idx += 256; n += 256; c += 256;
                if (n >= NH) { n -= NH; c += 1024 - NH; }
            }
        }
        __syncthreads();
        // (b)+(c) waves 0/1: cur2 = spk1 @ W2^T via WMMA, then LIF layer 2 + output
        if (wave < 2) {
            const int arow = wave * 16 + l16;
            v8f acc = {};
            for (int kk = 0; kk < 1024; kk += 32) {     // K padded with zeros past 1000
                v8h alo = *(const v8h*)&spk1h[arow * 1024 + kk + 16 * hh];
                v8h ahi = *(const v8h*)&spk1h[arow * 1024 + kk + 16 * hh + 8];
                v8h blo = *(const v8h*)&w2h[oo * 1024 + kk + 16 * hh];
                v8h bhi = *(const v8h*)&w2h[oo * 1024 + kk + 16 * hh + 8];
                v16h a, bb;
                #pragma unroll
                for (int i = 0; i < 8; ++i) {
                    a[i] = alo[i];  a[i + 8] = ahi[i];
                    bb[i] = blo[i]; bb[i + 8] = bhi[i];
                }
                acc = __builtin_amdgcn_wmma_f32_16x16x32_f16(false, a, false, bb,
                                                             (short)0, acc, false, false);
            }
            if (oo < NO) {
                #pragma unroll
                for (int r = 0; r < 8; ++r) {
                    float cur2  = acc[r] + b2v;
                    float m     = mem2r[r];
                    float reset = (m > THR) ? 1.0f : 0.0f;
                    float v     = BETA * m + cur2 - reset * THR;
                    float q     = (v - Q_LO) * (1.0f / Q_STEP);
                    q = fminf(fmaxf(q, 0.0f), Q_MAXI);
                    q = Q_LO + rintf(q) * Q_STEP;
                    mem2r[r] = q;
                    const float spk  = (q > THR) ? 1.0f : 0.0f;
                    const int   brow = b0 + wave * 16 + 8 * hh + r;
                    const size_t off = (size_t)t * B_TOT * NO + (size_t)brow * NO + oo;
                    out[off] = spk;                                     // spk2_rec
                    out[(size_t)T_STEPS * B_TOT * NO + off] = q;        // mem2_rec
                }
            }
        }
        __syncthreads();
    }
}

extern "C" void kernel_launch(void* const* d_in, const int* in_sizes, int n_in,
                              void* d_out, int out_size, void* d_ws, size_t ws_size,
                              hipStream_t stream)
{
    const float* x  = (const float*)d_in[0];
    const float* W1 = (const float*)d_in[1];
    const float* b1 = (const float*)d_in[2];
    const float* W2 = (const float*)d_in[3];
    const float* b2 = (const float*)d_in[4];
    float* out = (float*)d_out;

    _Float16* w1h = (_Float16*)((char*)d_ws + WS_W1H_OFF);   // 1.60 MB
    _Float16* xh  = (_Float16*)((char*)d_ws + WS_XH_OFF);    // 13.11 MB

    prep_f16<<<1024, 256, 0, stream>>>(x, W1, w1h, xh);

    // 292 KB dynamic LDS (> default cap) — raise the per-kernel limit. Host-side
    // attribute set; not a stream op, so safe (and deterministic) under capture.
    (void)hipFuncSetAttribute(reinterpret_cast<const void*>(snn_fused_wmma),
                              hipFuncAttributeMaxDynamicSharedMemorySize, LDS_BYTES);

    snn_fused_wmma<<<B_TOT / BT, 256, LDS_BYTES, stream>>>(xh, w1h, b1, W2, b2, out);
}
